// VQ_VAE_86681029968488
// MI455X (gfx1250) — matrix operand order repaired and checked by reading, not compiled, not run on hardware
//
#include <hip/hip_runtime.h>
#include <hip/hip_bf16.h>

#define BB 32
#define CC 256
#define HH 32
#define WW 32
#define HWP (HH * WW)          // 1024
#define NN (BB * HWP)          // 32768 query vectors
#define KK 1024                // codebooks

typedef __attribute__((ext_vector_type(16))) __bf16          v16bf;
typedef __attribute__((ext_vector_type(16))) unsigned short  v16u;
typedef __attribute__((ext_vector_type(8)))  float           v8f;

// ---- bf16 helpers (bit-level RNE, no reliance on __bf16 arithmetic) ----
__device__ __forceinline__ unsigned short f2bf(float f) {
    unsigned u = __float_as_uint(f);
    unsigned r = u + 0x7FFFu + ((u >> 16) & 1u);
    return (unsigned short)(r >> 16);
}
__device__ __forceinline__ float bf2f(unsigned short h) {
    return __uint_as_float(((unsigned)h) << 16);
}

// ============================================================================
// Kernel 1: permute x [B,C,H,W] -> ze [N,C]  (LDS tiled transpose per batch)
// block (32,8), grid (HW/32, C/32, B)
// ============================================================================
__global__ void vq_permute(const float* __restrict__ x, float* __restrict__ ze) {
    __shared__ float tile[32][33];
    const int b  = blockIdx.z;
    const int p0 = blockIdx.x * 32;   // hw tile
    const int c0 = blockIdx.y * 32;   // channel tile
    const int tx = threadIdx.x, ty = threadIdx.y;
    const float* xb = x + (size_t)b * CC * HWP;
#pragma unroll
    for (int i = 0; i < 4; ++i) {
        int c = c0 + ty + i * 8;
        tile[ty + i * 8][tx] = xb[(size_t)c * HWP + p0 + tx];   // coalesced over hw
    }
    __syncthreads();
    float* zb = ze + (size_t)b * HWP * CC;
#pragma unroll
    for (int i = 0; i < 4; ++i) {
        int p = p0 + ty + i * 8;
        zb[(size_t)p * CC + c0 + tx] = tile[tx][ty + i * 8];    // coalesced over c
    }
}

// ============================================================================
// Kernel 2: codebook -> bf16 hi/lo split + enorm[k] = ||cb_k||^2
// one block (256 threads) per codebook row
// ============================================================================
__global__ void vq_cbprep(const float* __restrict__ cb,
                          unsigned short* __restrict__ cb_hi,
                          unsigned short* __restrict__ cb_lo,
                          float* __restrict__ enorm) {
    const int k = blockIdx.x;
    const int c = threadIdx.x;
    float v = cb[(size_t)k * CC + c];
    unsigned short h = f2bf(v);
    cb_hi[(size_t)k * CC + c] = h;
    cb_lo[(size_t)k * CC + c] = f2bf(v - bf2f(h));
    float s = v * v;
#pragma unroll
    for (int off = 16; off; off >>= 1) s += __shfl_down(s, off, 32);
    __shared__ float red[8];
    if ((threadIdx.x & 31) == 0) red[threadIdx.x >> 5] = s;
    __syncthreads();
    if (threadIdx.x < 8) {
        float t = red[threadIdx.x];
        t += __shfl_down(t, 4, 8);
        t += __shfl_down(t, 2, 8);
        t += __shfl_down(t, 1, 8);
        if (threadIdx.x == 0) enorm[k] = t;
    }
}

// ============================================================================
// Kernel 3: split-precision bf16 WMMA scoring + argmin
// one wave per 16 rows of ze; loop over 64 tiles of 16 codebooks
// score = cb_hi*ze_hi + cb_lo*ze_hi + cb_hi*ze_lo  (3x bf16 ~ fp32 accuracy)
// dist_rel = enorm[k] - 2*score  (||z||^2 constant per row -> irrelevant)
// launch_bounds(256, 2): budget VGPRs for 2 waves/EU so Ah+Al (128 VGPRs)
// stay resident -- previous build spilled Al to scratch inside the hot loop.
// Three independent accumulator chains (hh/hl/lh) give WMMA ILP at the
// reduced occupancy.
// ============================================================================
__global__ void __launch_bounds__(256, 2)
vq_score(const float* __restrict__ ze,
         const unsigned short* __restrict__ cb_hi,
         const unsigned short* __restrict__ cb_lo,
         const float* __restrict__ enorm,
         int* __restrict__ nearest) {
    const int lane = threadIdx.x & 31;
    const int wave = blockIdx.x * (blockDim.x >> 5) + (threadIdx.x >> 5);
    const int n0   = wave * 16;         // this wave's 16 query rows
    const int half = lane >> 4;         // 0 / 1
    const int lrow = lane & 15;

    // ---- Build A fragments (16x32 bf16 per chunk), hi & lo, from fp32 ze ----
    // A layout: lane<16 holds row M=lane, K=0..15 contiguous;
    //           lane>=16 holds row M=lane-16, K=16..31 contiguous.
    v16bf Ah[8], Al[8];
    const float* zrow = ze + (size_t)(n0 + lrow) * CC;
#pragma unroll
    for (int ch = 0; ch < 8; ++ch) {
        const int c0 = ch * 32 + half * 16;
        v16u h, l;
#pragma unroll
        for (int j = 0; j < 8; ++j) {
            float v0 = zrow[c0 + j];
            float v1 = zrow[c0 + 8 + j];
            unsigned short h0 = f2bf(v0), h1 = f2bf(v1);
            h[j] = h0;                 h[8 + j] = h1;
            l[j] = f2bf(v0 - bf2f(h0)); l[8 + j] = f2bf(v1 - bf2f(h1));
        }
        Ah[ch] = __builtin_bit_cast(v16bf, h);
        Al[ch] = __builtin_bit_cast(v16bf, l);
    }

    float best[8];
    int   bidx[8];
#pragma unroll
    for (int i = 0; i < 8; ++i) { best[i] = 3.4e38f; bidx[i] = 0; }

    for (int kt = 0; kt < KK / 16; ++kt) {
        const int myk = kt * 16 + lrow;   // codebook column this lane owns in D
        // B layout (32x16 bf16): lane<16 -> col N=lane, K=0..15 contiguous;
        //                        lane>=16 -> col N=lane-16, K=16..31 contiguous.
        const unsigned short* bh = cb_hi + (size_t)myk * CC + half * 16;
        const unsigned short* bl = cb_lo + (size_t)myk * CC + half * 16;
        if (kt + 1 < KK / 16) {   // prefetch next tile (global_prefetch_b8)
            __builtin_prefetch(cb_hi + (size_t)(myk + 16) * CC, 0, 1);
            __builtin_prefetch(cb_lo + (size_t)(myk + 16) * CC, 0, 1);
        }
        v8f acc_hh = {0.f, 0.f, 0.f, 0.f, 0.f, 0.f, 0.f, 0.f};
        v8f acc_hl = {0.f, 0.f, 0.f, 0.f, 0.f, 0.f, 0.f, 0.f};
        v8f acc_lh = {0.f, 0.f, 0.f, 0.f, 0.f, 0.f, 0.f, 0.f};
#pragma unroll
        for (int ch = 0; ch < 8; ++ch) {
            v16bf Bh = __builtin_bit_cast(v16bf, *(const v16u*)(bh + ch * 32));
            v16bf Bl = __builtin_bit_cast(v16bf, *(const v16u*)(bl + ch * 32));
            acc_hh = __builtin_amdgcn_wmma_f32_16x16x32_bf16(false, Ah[ch], false, Bh,
                                                             (short)0, acc_hh, false, false);
            acc_hl = __builtin_amdgcn_wmma_f32_16x16x32_bf16(false, Ah[ch], false, Bl,
                                                             (short)0, acc_hl, false, false);
            acc_lh = __builtin_amdgcn_wmma_f32_16x16x32_bf16(false, Al[ch], false, Bh,
                                                             (short)0, acc_lh, false, false);
        }
        const float en = enorm[myk];
        // D layout: vgpr i, lanes 0-15 -> row M=i ; lanes 16-31 -> row M=8+i
#pragma unroll
        for (int i = 0; i < 8; ++i) {
            float s = acc_hh[i] + (acc_hl[i] + acc_lh[i]);
            float d = en - 2.0f * s;
            if (d < best[i]) { best[i] = d; bidx[i] = myk; }
        }
    }

    // Cross-lane min+argmin over the 16 lanes that share each row set
#pragma unroll
    for (int off = 1; off < 16; off <<= 1) {
#pragma unroll
        for (int i = 0; i < 8; ++i) {
            float ov = __shfl_xor(best[i], off, 32);
            int   oi = __shfl_xor(bidx[i], off, 32);
            if (ov < best[i] || (ov == best[i] && oi < bidx[i])) {
                best[i] = ov; bidx[i] = oi;
            }
        }
    }
    if (lrow == 0) {                       // lane 0 -> rows 0..7, lane 16 -> rows 8..15
#pragma unroll
        for (int i = 0; i < 8; ++i)
            nearest[n0 + half * 8 + i] = bidx[i];
    }
}

// ============================================================================
// Kernel 4: gather exact fp32 codebook rows -> e [N,C] and quantized [B,C,H,W]
// block (32,8), grid (HW/32, C/32, B); LDS transpose for the [B,C,H,W] write
// ============================================================================
__global__ void vq_gather(const float* __restrict__ cb,
                          const int* __restrict__ nearest,
                          float* __restrict__ out_q,
                          float* __restrict__ out_e) {
    __shared__ float tile[32][33];
    __shared__ int idx[32];
    const int b  = blockIdx.z;
    const int p0 = blockIdx.x * 32;
    const int c0 = blockIdx.y * 32;
    const int tx = threadIdx.x, ty = threadIdx.y;
    if (ty == 0) idx[tx] = nearest[b * HWP + p0 + tx];
    __syncthreads();
#pragma unroll
    for (int i = 0; i < 4; ++i) {
        int pl = ty + i * 8;
        float v = cb[(size_t)idx[pl] * CC + c0 + tx];           // coalesced over c
        tile[pl][tx] = v;
        out_e[(size_t)(b * HWP + p0 + pl) * CC + c0 + tx] = v;  // coalesced over c
    }
    __syncthreads();
#pragma unroll
    for (int i = 0; i < 4; ++i) {
        int cl = ty + i * 8;
        out_q[((size_t)b * CC + c0 + cl) * HWP + p0 + tx] = tile[tx][cl]; // coalesced over hw
    }
}

// ============================================================================
extern "C" void kernel_launch(void* const* d_in, const int* in_sizes, int n_in,
                              void* d_out, int out_size, void* d_ws, size_t ws_size,
                              hipStream_t stream) {
    const float* x  = (const float*)d_in[0];   // [B,C,H,W]
    const float* cb = (const float*)d_in[1];   // [K,C]

    float* out_q  = (float*)d_out;                       // [B,C,H,W]
    float* out_ze = out_q  + (size_t)BB * CC * HWP;      // [N,C]
    float* out_e  = out_ze + (size_t)NN * CC;            // [N,C]

    // workspace layout (~1.2 MB)
    char* ws = (char*)d_ws;
    unsigned short* cb_hi = (unsigned short*)ws;  ws += (size_t)KK * CC * sizeof(unsigned short);
    unsigned short* cb_lo = (unsigned short*)ws;  ws += (size_t)KK * CC * sizeof(unsigned short);
    float* enorm = (float*)ws;                    ws += (size_t)KK * sizeof(float);
    int*   nearest = (int*)ws;                    ws += (size_t)NN * sizeof(int);

    dim3 tgrid(HWP / 32, CC / 32, BB), tblk(32, 8);
    vq_permute<<<tgrid, tblk, 0, stream>>>(x, out_ze);
    vq_cbprep<<<KK, CC, 0, stream>>>(cb, cb_hi, cb_lo, enorm);
    // 2048 waves, 8 waves (256 threads) per block
    vq_score<<<(NN / 16) / 8, 256, 0, stream>>>(out_ze, cb_hi, cb_lo, enorm, nearest);
    vq_gather<<<tgrid, tblk, 0, stream>>>(cb, nearest, out_q, out_e);
}
